// EHRCPCCLoss_51307679318816
// MI455X (gfx1250) — matrix operand order repaired and checked by reading, not compiled, not run on hardware
//
#include <hip/hip_runtime.h>
#include <hip/hip_bf16.h>
#include <math.h>

typedef __attribute__((ext_vector_type(16))) __bf16 v16bf;
typedef __attribute__((ext_vector_type(8)))  __bf16 v8bf;
typedef __attribute__((ext_vector_type(8)))  float  v8f;

#define BDIM 4096
#define DDIM 256
#define TILE 128
#define KCH  32
#define PITCH 40   // 32 + 8 halves padding (80B row stride, 16B aligned)

#ifndef USE_ASYNC_COPY
#define USE_ASYNC_COPY 1   // gfx1250 GLOBAL_LOAD_ASYNC_TO_LDS_B128 path
#endif

// ---------- helpers: explicit RNE f32<->bf16 (no reliance on __bf16 arithmetic) ----------
__device__ __forceinline__ __bf16 f32_to_bf16(float x) {
    union { float f; unsigned u; } c; c.f = x;
    unsigned r = c.u + 0x7FFFu + ((c.u >> 16) & 1u);   // round-to-nearest-even
    unsigned short hv = (unsigned short)(r >> 16);
    __bf16 out;
    __builtin_memcpy(&out, &hv, 2);
    return out;
}
__device__ __forceinline__ float bf16_to_f32(__bf16 h) {
    unsigned short hv; __builtin_memcpy(&hv, &h, 2);
    union { unsigned u; float f; } c; c.u = ((unsigned)hv) << 16;
    return c.f;
}
__device__ __forceinline__ v16bf make_frag(const __bf16* p0, const __bf16* p1) {
    v8bf a = *(const v8bf*)p0;
    v8bf b = *(const v8bf*)p1;
    v16bf r;
#pragma unroll
    for (int i = 0; i < 8; ++i) { r[i] = a[i]; r[8 + i] = b[i]; }
    return r;
}

// ---------- 32-byte global->LDS copy (async on CDNA5; ASYNCcnt tracked, in-order) ----------
__device__ __forceinline__ void copy32(const __bf16* g, __bf16* l) {
#if USE_ASYNC_COPY
    // Low 32 bits of the generic shared-variable address are the LDS byte offset.
    unsigned laddr = (unsigned)(size_t)l;
    // each moves 16 bytes from g (+offset) to LDS byte address laddr (+offset)
    asm volatile("global_load_async_to_lds_b128 %0, %1, off"
                 :: "v"(laddr), "v"(g) : "memory");
    asm volatile("global_load_async_to_lds_b128 %0, %1, off offset:16"
                 :: "v"(laddr), "v"(g) : "memory");
#else
    const uint4* s = (const uint4*)g;
    uint4* d = (uint4*)l;
    d[0] = s[0];
    d[1] = s[1];
#endif
}
// Wait until at most N async ops remain outstanding (in-order completion).
__device__ __forceinline__ void async_wait_le8() {
#if USE_ASYNC_COPY
    asm volatile("s_wait_asynccnt 0x8" ::: "memory");
#endif
}
__device__ __forceinline__ void async_wait_le0() {
#if USE_ASYNC_COPY
    asm volatile("s_wait_asynccnt 0x0" ::: "memory");
#endif
}

// ---------- kernel 1: bf16x2 split of R + row squared norms + zero fp64 accumulators ----------
__global__ __launch_bounds__(256) void prep_kernel(const float* __restrict__ R,
                                                   __bf16* __restrict__ Rhi,
                                                   __bf16* __restrict__ Rlo,
                                                   float* __restrict__ sq,
                                                   double* __restrict__ acc) {
    __shared__ float red[256];
    const int row = blockIdx.x;
    const int t = threadIdx.x;
    const size_t idx = (size_t)row * DDIM + t;
    float v = R[idx];
    __bf16 h = f32_to_bf16(v);
    Rhi[idx] = h;
    Rlo[idx] = f32_to_bf16(v - bf16_to_f32(h));
    red[t] = v * v;
    __syncthreads();
    for (int s = 128; s > 0; s >>= 1) {
        if (t < s) red[t] += red[t + s];
        __syncthreads();
    }
    if (t == 0) sq[row] = red[0];
    if (blockIdx.x == 0 && t < 4) acc[t] = 0.0;
}

// ---------- kernel 2: fused Gram (bf16x2 WMMA) + distance epilogue + stats ----------
__global__ __launch_bounds__(256) void gram_corr_kernel(const __bf16* __restrict__ Rhi,
                                                        const __bf16* __restrict__ Rlo,
                                                        const int* __restrict__ lab,
                                                        const float* __restrict__ sq,
                                                        double* __restrict__ acc) {
    const int ti = blockIdx.y;
    const int tj = blockIdx.x;
    if (tj < ti) return;                       // upper-triangle blocks only (uniform exit)

    // Double-buffered staging: 2 x 4 arrays x 10240B = 80KB LDS
    __shared__ __bf16 Ahi[2][TILE][PITCH], Alo[2][TILE][PITCH];
    __shared__ __bf16 Bhi[2][TILE][PITCH], Blo[2][TILE][PITCH];

    const int t    = threadIdx.x;
    const int lane = t & 31;
    const int wave = t >> 5;
    const int wm   = wave & 3;   // 4 waves along M (4*32 = 128)
    const int wn   = wave >> 2;  // 2 waves along N (2*64 = 128)

    v8f accm[2][4];
#pragma unroll
    for (int mt = 0; mt < 2; ++mt)
#pragma unroll
        for (int nt = 0; nt < 4; ++nt) accm[mt][nt] = (v8f){};

    const int ldrow = t >> 1;            // 0..127
    const int ldoff = (t & 1) * 16;      // 0 or 16 halves (32B per lane per array)
    const size_t gaBase = (size_t)(ti * TILE + ldrow) * DDIM + ldoff;
    const size_t gbBase = (size_t)(tj * TILE + ldrow) * DDIM + ldoff;

    // prologue: stage chunk 0 into buffer 0 (8 async b128 per lane)
    copy32(Rhi + gaBase, &Ahi[0][ldrow][ldoff]);
    copy32(Rlo + gaBase, &Alo[0][ldrow][ldoff]);
    copy32(Rhi + gbBase, &Bhi[0][ldrow][ldoff]);
    copy32(Rlo + gbBase, &Blo[0][ldrow][ldoff]);

    for (int kc = 0; kc < DDIM / KCH; ++kc) {
        const int cur = kc & 1;
        const int nxt = cur ^ 1;
        if (kc < DDIM / KCH - 1) {
            // issue next chunk's copies into the other buffer, then wait for
            // the current chunk only (ASYNCcnt completes in order: 16 -> 8).
            const int ko = (kc + 1) * KCH;
            copy32(Rhi + gaBase + ko, &Ahi[nxt][ldrow][ldoff]);
            copy32(Rlo + gaBase + ko, &Alo[nxt][ldrow][ldoff]);
            copy32(Rhi + gbBase + ko, &Bhi[nxt][ldrow][ldoff]);
            copy32(Rlo + gbBase + ko, &Blo[nxt][ldrow][ldoff]);
            async_wait_le8();
        } else {
            async_wait_le0();
        }
        __syncthreads();       // all waves have chunk kc in LDS

        // ---- WMMA compute: D = A x B + C, bf16x2 (hi*hi + hi*lo + lo*hi) ----
        const int abase = (lane < 16) ? 0 : 16;   // A 16x32 bf16 per-lane K layout
        const int bbase = (lane < 16) ? 0 : 16;   // B 32x16 bf16 per-lane K layout
#pragma unroll
        for (int mt = 0; mt < 2; ++mt) {
            const int ar = wm * 32 + mt * 16 + (lane & 15);
            v16bf fahi = make_frag(&Ahi[cur][ar][abase], &Ahi[cur][ar][abase + 8]);
            v16bf falo = make_frag(&Alo[cur][ar][abase], &Alo[cur][ar][abase + 8]);
#pragma unroll
            for (int nt = 0; nt < 4; ++nt) {
                const int br = wn * 64 + nt * 16 + (lane & 15);
                v16bf fbhi = make_frag(&Bhi[cur][br][bbase], &Bhi[cur][br][bbase + 8]);
                v16bf fblo = make_frag(&Blo[cur][br][bbase], &Blo[cur][br][bbase + 8]);
                accm[mt][nt] = __builtin_amdgcn_wmma_f32_16x16x32_bf16(
                    false, fahi, false, fbhi, (short)0, accm[mt][nt], false, false);
                accm[mt][nt] = __builtin_amdgcn_wmma_f32_16x16x32_bf16(
                    false, fahi, false, fblo, (short)0, accm[mt][nt], false, false);
                accm[mt][nt] = __builtin_amdgcn_wmma_f32_16x16x32_bf16(
                    false, falo, false, fbhi, (short)0, accm[mt][nt], false, false);
            }
        }
        __syncthreads();       // everyone done reading buf[cur]; safe to overwrite
    }

    // ---- epilogue: distances + correlation statistics over upper-triangle pairs ----
    float sx = 0.f, sxx = 0.f, sxy = 0.f, syc = 0.f;
#pragma unroll
    for (int mt = 0; mt < 2; ++mt) {
#pragma unroll
        for (int nt = 0; nt < 4; ++nt) {
            const int col = tj * TILE + wn * 64 + nt * 16 + (lane & 15);
            const float sqc = sq[col];
            const int lc = lab[col];
#pragma unroll
            for (int r = 0; r < 8; ++r) {
                const int row = ti * TILE + wm * 32 + mt * 16 + ((lane < 16) ? r : r + 8);
                if (col > row) {
                    float g  = accm[mt][nt][r];
                    float d2 = sq[row] + sqc - 2.0f * g;
                    d2 = fmaxf(d2, 0.0f);
                    float x = sqrtf(d2);
                    sx  += x;
                    sxx += d2;
                    if (lab[row] != lc) { syc += 1.0f; sxy += x; }
                }
            }
        }
    }
    // wave32 reduction
#pragma unroll
    for (int off = 16; off > 0; off >>= 1) {
        sx  += __shfl_xor(sx,  off, 32);
        sxx += __shfl_xor(sxx, off, 32);
        sxy += __shfl_xor(sxy, off, 32);
        syc += __shfl_xor(syc, off, 32);
    }
    if (lane == 0) {
        atomicAdd(&acc[0], (double)sx);
        atomicAdd(&acc[1], (double)sxx);
        atomicAdd(&acc[2], (double)sxy);
        atomicAdd(&acc[3], (double)syc);
    }
}

// ---------- kernel 3: scalar finish ----------
__global__ void finalize_kernel(const double* __restrict__ acc, float* __restrict__ out) {
    const double P = (double)BDIM * (double)(BDIM - 1) * 0.5;  // 8386560
    double Sx = acc[0], Sxx = acc[1], Sxy = acc[2], Sy = acc[3];
    double num = Sxy - Sx * Sy / P;
    double dx  = Sxx - Sx * Sx / P;
    double dy  = Sy  - Sy * Sy / P;     // y in {0,1} => Syy == Sy
    double corr = num / sqrt(dx * dy);
    double loss = 1.0 - corr;
    if (!(loss == loss)) loss = 1.0;    // NaN guard
    out[0] = (float)loss;
}

extern "C" void kernel_launch(void* const* d_in, const int* in_sizes, int n_in,
                              void* d_out, int out_size, void* d_ws, size_t ws_size,
                              hipStream_t stream) {
    const float* R  = (const float*)d_in[0];
    const int* lab  = (const int*)d_in[1];

    // workspace layout: [0,32) fp64 accumulators; [256,16640) sq; [32768, +4MB) bf16 hi/lo
    double* acc = (double*)d_ws;
    float*  sq  = (float*)((char*)d_ws + 256);
    __bf16* Rhi = (__bf16*)((char*)d_ws + 32768);
    __bf16* Rlo = Rhi + (size_t)BDIM * DDIM;

    prep_kernel<<<BDIM, 256, 0, stream>>>(R, Rhi, Rlo, sq, acc);

    dim3 grid(BDIM / TILE, BDIM / TILE);               // tj = x, ti = y; tj<ti exits
    gram_corr_kernel<<<grid, 256, 0, stream>>>(Rhi, Rlo, lab, sq, acc);

    finalize_kernel<<<1, 1, 0, stream>>>(acc, (float*)d_out);
}
